// STGNNLayer_26018911879573
// MI455X (gfx1250) — matrix operand order repaired and checked by reading, not compiled, not run on hardware
//
#include <hip/hip_runtime.h>
#include <hip/hip_bf16.h>
#include <math.h>

// B=8, C=64, N=512, T=24, H=4, D=16, C8=8, Cf=256
// R = B*N*T = 98304 rows in [*,C] "ntc" layout

typedef __attribute__((ext_vector_type(16))) _Float16 v16h;
typedef __attribute__((ext_vector_type(8)))  float    v8f;
typedef __attribute__((address_space(3))) const void lds_cv;

__device__ __forceinline__ unsigned lds_addr_of(const void* p) {
  return (unsigned)(uintptr_t)(lds_cv*)p;
}
// one 16-byte async DMA transfer per lane: global (f16 data) -> LDS (ASYNCcnt)
__device__ __forceinline__ void async_b128(const void* gsrc, unsigned lds_dst) {
  asm volatile("global_load_async_to_lds_b128 %0, %1, off"
               :: "v"(lds_dst), "v"((unsigned long long)(uintptr_t)gsrc)
               : "memory");
}
__device__ __forceinline__ void wait_async0() {
  asm volatile("s_wait_asynccnt 0x0" ::: "memory");
}
__device__ __forceinline__ void wait_async1() {
  asm volatile("s_wait_asynccnt 0x1" ::: "memory");
}
__device__ __forceinline__ void wait_async2() {
  asm volatile("s_wait_asynccnt 0x2" ::: "memory");
}

__device__ __forceinline__ float wave_max(float v) {
#pragma unroll
  for (int d = 16; d; d >>= 1) v = fmaxf(v, __shfl_xor(v, d, 32));
  return v;
}
__device__ __forceinline__ float wave_sum(float v) {
#pragma unroll
  for (int d = 16; d; d >>= 1) v += __shfl_xor(v, d, 32);
  return v;
}

// A fragment: 16x32 f16, lane holds row (lane&15); K contiguous: k = 16*(lane>>4) + s
__device__ __forceinline__ v16h frag_a(const _Float16* base, int stride, int lane) {
  const int r = lane & 15;
  const int off = (lane >> 4) << 4;
  const _Float16* p = base + r * stride + off;
  v16h f;
#pragma unroll
  for (int h = 0; h < 16; ++h) f[h] = p[h];
  return f;
}
// B fragment: 32x16 f16 from N-major (transposed) storage Bt[n][k]
__device__ __forceinline__ v16h frag_b(const _Float16* base, int stride, int lane) {
  const int n = lane & 15;
  const int off = (lane >> 4) << 4;
  const _Float16* p = base + n * stride + off;
  v16h f;
#pragma unroll
  for (int h = 0; h < 16; ++h) f[h] = p[h];
  return f;
}
__device__ __forceinline__ v8f wmma32(v16h a, v16h b, v8f c) {
  return __builtin_amdgcn_wmma_f32_16x16x32_f16(false, a, false, b, (short)0, c,
                                                false, false);
}

// ---------------- K1: spatial 1x1 convs q,k,v  ([B,Cout,T,N] layouts) --------
__global__ void k_spatial_qkv(const float* __restrict__ x,
                              const float* __restrict__ wq_s, const float* __restrict__ bq_s,
                              const float* __restrict__ wk_s, const float* __restrict__ bk_s,
                              const float* __restrict__ wv_s, const float* __restrict__ bv_s,
                              float* __restrict__ qc, float* __restrict__ kc,
                              _Float16* __restrict__ vch) {
  __shared__ float xl[64][128];
  const int bt = blockIdx.x;
  const int b = bt / 24, t = bt % 24;
  const int n0 = blockIdx.y * 128;
  for (int idx = threadIdx.x; idx < 64 * 128; idx += 256) {
    int c = idx >> 7, nn = idx & 127;
    xl[c][nn] = x[((b * 64 + c) * 512 + n0 + nn) * 24 + t];
  }
  __syncthreads();
  for (int idx = threadIdx.x; idx < 80 * 128; idx += 256) {
    int co = idx >> 7, nn = idx & 127;
    const float* w; float bias;
    if (co < 8)       { w = wq_s + co * 64;        bias = bq_s[co]; }
    else if (co < 16) { w = wk_s + (co - 8) * 64;  bias = bk_s[co - 8]; }
    else              { w = wv_s + (co - 16) * 64; bias = bv_s[co - 16]; }
    float acc = bias;
#pragma unroll 8
    for (int c = 0; c < 64; ++c) acc += w[c] * xl[c][nn];
    if (co < 8)
      qc[((b * 8 + co) * 24 + t) * 512 + n0 + nn] = acc;
    else if (co < 16)
      kc[((b * 8 + (co - 8)) * 24 + t) * 512 + n0 + nn] = acc;
    else
      vch[((b * 64 + (co - 16)) * 24 + t) * 512 + n0 + nn] = (_Float16)acc;
  }
}

// ---------------- K2: spatial scores + softmax + adj-mask + re-softmax -------
__global__ void k_spatial_attn(const float* __restrict__ qc, const float* __restrict__ kc,
                               const float* __restrict__ adj, float* __restrict__ attn,
                               _Float16* __restrict__ attn16) {
  __shared__ float Kl[8][512];
  __shared__ float Sl[8][512];
  const int bt = blockIdx.x;
  const int i0 = blockIdx.y * 8;
  for (int idx = threadIdx.x; idx < 8 * 512; idx += 256) {
    int c8 = idx >> 9, j = idx & 511;
    Kl[c8][j] = kc[(bt * 8 + c8) * 512 + j];
  }
  __syncthreads();
  const int w = threadIdx.x >> 5, lane = threadIdx.x & 31;
  const int i = i0 + w;
  float qv[8];
#pragma unroll
  for (int c8 = 0; c8 < 8; ++c8) qv[c8] = qc[(bt * 8 + c8) * 512 + i];
  float lmax = -3.4e38f;
  for (int j = lane; j < 512; j += 32) {
    float s = 0.f;
#pragma unroll
    for (int c8 = 0; c8 < 8; ++c8) s += qv[c8] * Kl[c8][j];
    s *= 0.125f;                        // 1/sqrt(C)
    Sl[w][j] = s;
    lmax = fmaxf(lmax, s);
  }
  float m = wave_max(lmax);
  float lsum = 0.f;
  for (int j = lane; j < 512; j += 32) {
    float e = __expf(Sl[w][j] - m);
    Sl[w][j] = e;
    lsum += e;
  }
  float inv = 1.f / wave_sum(lsum);
  float lmax2 = -3.4e38f;
  for (int j = lane; j < 512; j += 32) {
    float p = Sl[w][j] * inv;
    Sl[w][j] = p;
    if (adj[i * 512 + j] != 0.f) lmax2 = fmaxf(lmax2, p);
  }
  float m2 = wave_max(lmax2);
  float lsum2 = 0.f;
  for (int j = lane; j < 512; j += 32) {
    float e = (adj[i * 512 + j] != 0.f) ? __expf(Sl[w][j] - m2) : 0.f;
    Sl[w][j] = e;
    lsum2 += e;
  }
  float inv2 = 1.f / wave_sum(lsum2);
  const size_t rowbase = (size_t)(bt * 512 + i) * 512;
  for (int j = lane; j < 512; j += 32) {
    float a = Sl[w][j] * inv2;
    attn[rowbase + j] = a;
    attn16[rowbase + j] = (_Float16)a;
  }
}

// ------ K3: xs = V @ attn^T  (WMMA, K=512, double-buffered async staging) ----
__global__ void k_spatial_av(const _Float16* __restrict__ vch,
                             const _Float16* __restrict__ attn16,
                             float* __restrict__ xs) {
  __shared__ _Float16 Al[2][64][32];
  __shared__ _Float16 Bl[2][64][32];
  const int bt = blockIdx.y;
  const int i0 = blockIdx.x * 64;
  const int w = threadIdx.x >> 5, lane = threadIdx.x & 31;
  const int sr = threadIdx.x >> 2;          // staging row 0..63
  const int sp = (threadIdx.x & 3) << 3;    // staging half-offset 0,8,16,24
  const unsigned ldsA[2] = { lds_addr_of(&Al[0][sr][sp]), lds_addr_of(&Al[1][sr][sp]) };
  const unsigned ldsB[2] = { lds_addr_of(&Bl[0][sr][sp]), lds_addr_of(&Bl[1][sr][sp]) };
  const _Float16* gA = vch + ((bt * 64 + sr) * 512 + sp);
  const _Float16* gB = attn16 + ((size_t)(bt * 512 + i0 + sr) * 512 + sp);
  v8f acc[2] = {};
  // prologue: stream tile 0 while nothing is in flight
  async_b128(gA, ldsA[0]);
  async_b128(gB, ldsB[0]);
  for (int ks = 0; ks < 16; ++ks) {
    const int bf = ks & 1;
    if (ks < 15) {           // prefetch next tile into the other buffer
      async_b128(gA + (ks + 1) * 32, ldsA[bf ^ 1]);
      async_b128(gB + (ks + 1) * 32, ldsB[bf ^ 1]);
      wait_async2();         // in-order completion: oldest 2 (tile ks) are done
    } else {
      wait_async0();
    }
    __syncthreads();
#pragma unroll
    for (int tt = 0; tt < 2; ++tt) {
      int tile = w + tt * 8;
      int mt = tile >> 2, nt = tile & 3;
      v16h a = frag_a(&Al[bf][mt * 16][0], 32, lane);
      v16h bfrag = frag_b(&Bl[bf][nt * 16][0], 32, lane);
      acc[tt] = wmma32(a, bfrag, acc[tt]);
    }
    __syncthreads();         // all reads done before this buffer is refilled
  }
  const int rr = (lane >> 4) << 3, col = lane & 15;
#pragma unroll
  for (int tt = 0; tt < 2; ++tt) {
    int tile = w + tt * 8;
    int mt = tile >> 2, nt = tile & 3;
#pragma unroll
    for (int v = 0; v < 8; ++v)
      xs[(bt * 64 + mt * 16 + rr + v) * 512 + i0 + nt * 16 + col] = acc[tt][v];
  }
}

// ---------------- K4: residual + LN1 over C; relayout to [B,N,T,C] -----------
__global__ void k_res_ln1(const float* __restrict__ x, const float* __restrict__ xs,
                          const float* __restrict__ g, const float* __restrict__ bta,
                          float* __restrict__ x1t) {
  __shared__ float s[24][64];
  __shared__ float mu[24], rs[24];
  const int b = blockIdx.x >> 9;
  const int n = blockIdx.x & 511;
  for (int idx = threadIdx.x; idx < 1536; idx += 256) {
    int c = idx / 24, t = idx % 24;
    s[t][c] = x[((b * 64 + c) * 512 + n) * 24 + t] +
              xs[((b * 24 + t) * 64 + c) * 512 + n];
  }
  __syncthreads();
  if (threadIdx.x < 24) {
    int t = threadIdx.x;
    float sm = 0.f, sq = 0.f;
    for (int c = 0; c < 64; ++c) { float v = s[t][c]; sm += v; sq += v * v; }
    float m = sm * (1.f / 64.f);
    float var = sq * (1.f / 64.f) - m * m;
    mu[t] = m; rs[t] = rsqrtf(var + 1e-5f);
  }
  __syncthreads();
  for (int idx = threadIdx.x; idx < 1536; idx += 256) {
    int t = idx >> 6, c = idx & 63;
    x1t[((b * 512 + n) * 24 + t) * 64 + c] = (s[t][c] - mu[t]) * rs[t] * g[c] + bta[c];
  }
}

// ---------------- K5: temporal Q/K/V projections (WMMA) ----------------------
__global__ void k_tqkv(const float* __restrict__ x1t,
                       const float* __restrict__ wq, const float* __restrict__ bq,
                       const float* __restrict__ wk, const float* __restrict__ bk,
                       const float* __restrict__ wv, const float* __restrict__ bv,
                       float* __restrict__ Qt, float* __restrict__ Kt,
                       float* __restrict__ Vt) {
  __shared__ _Float16 Al[64][66];
  __shared__ _Float16 Wl[3][64][66];
  const int r0 = blockIdx.x * 64;
  for (int idx = threadIdx.x; idx < 4096; idx += 256) {
    int o = idx >> 6, c = idx & 63;
    Wl[0][o][c] = (_Float16)wq[c * 64 + o];
    Wl[1][o][c] = (_Float16)wk[c * 64 + o];
    Wl[2][o][c] = (_Float16)wv[c * 64 + o];
    Al[o][c]    = (_Float16)x1t[(r0 + o) * 64 + c];
  }
  __syncthreads();
  const int w = threadIdx.x >> 5, lane = threadIdx.x & 31;
  const int rr = (lane >> 4) << 3, col = lane & 15;
  for (int tile = w; tile < 48; tile += 8) {
    int mat = tile >> 4;
    int rem = tile & 15;
    int mt = rem >> 2, nt = rem & 3;
    v8f acc = {};
#pragma unroll
    for (int ks = 0; ks < 2; ++ks) {
      v16h a = frag_a(&Al[mt * 16][ks * 32], 66, lane);
      v16h bf = frag_b(&Wl[mat][nt * 16][ks * 32], 66, lane);
      acc = wmma32(a, bf, acc);
    }
    float* outp = (mat == 0) ? Qt : (mat == 1) ? Kt : Vt;
    const float* bias = (mat == 0) ? bq : (mat == 1) ? bk : bv;
    int o = nt * 16 + col;
    float bsv = bias[o];
#pragma unroll
    for (int v = 0; v < 8; ++v)
      outp[(r0 + mt * 16 + rr + v) * 64 + o] = acc[v] + bsv;
  }
}

// ---------------- K6: temporal multi-head attention over T=24 ----------------
__global__ void k_tattn(const float* __restrict__ Qt, const float* __restrict__ Kt,
                        const float* __restrict__ Vt, float* __restrict__ att) {
  __shared__ float q[24][64], k[24][64], v[24][64];
  const int rbase = blockIdx.x * 24;
  for (int idx = threadIdx.x; idx < 1536; idx += 128) {
    int t = idx >> 6, c = idx & 63;
    q[t][c] = Qt[(rbase + t) * 64 + c];
    k[t][c] = Kt[(rbase + t) * 64 + c];
    v[t][c] = Vt[(rbase + t) * 64 + c];
  }
  __syncthreads();
  if (threadIdx.x < 96) {
    int h = threadIdx.x / 24, t = threadIdx.x % 24;
    float sc[24], m = -3.4e38f;
    for (int s2 = 0; s2 < 24; ++s2) {
      float d = 0.f;
#pragma unroll
      for (int dd = 0; dd < 16; ++dd) d += q[t][h * 16 + dd] * k[s2][h * 16 + dd];
      d *= 0.25f;                      // 1/sqrt(D)
      sc[s2] = d;
      m = fmaxf(m, d);
    }
    float sum = 0.f;
    for (int s2 = 0; s2 < 24; ++s2) { sc[s2] = __expf(sc[s2] - m); sum += sc[s2]; }
    float inv = 1.f / sum;
    for (int dd = 0; dd < 16; ++dd) {
      float o = 0.f;
      for (int s2 = 0; s2 < 24; ++s2) o += sc[s2] * v[s2][h * 16 + dd];
      att[(rbase + t) * 64 + h * 16 + dd] = o * inv;
    }
  }
}

// ---------------- K7: out-projection + residual (WMMA) -----------------------
__global__ void k_wo_res(const float* __restrict__ att, const float* __restrict__ x1t,
                         const float* __restrict__ wo, const float* __restrict__ bo,
                         float* __restrict__ x2t) {
  __shared__ _Float16 Al[64][66];
  __shared__ _Float16 Wl[64][66];
  const int r0 = blockIdx.x * 64;
  for (int idx = threadIdx.x; idx < 4096; idx += 256) {
    int o = idx >> 6, c = idx & 63;
    Wl[o][c] = (_Float16)wo[c * 64 + o];
    Al[o][c] = (_Float16)att[(r0 + o) * 64 + c];
  }
  __syncthreads();
  const int w = threadIdx.x >> 5, lane = threadIdx.x & 31;
  const int rr = (lane >> 4) << 3, col = lane & 15;
  for (int tile = w; tile < 16; tile += 8) {
    int mt = tile >> 2, nt = tile & 3;
    v8f acc = {};
#pragma unroll
    for (int ks = 0; ks < 2; ++ks) {
      v16h a = frag_a(&Al[mt * 16][ks * 32], 66, lane);
      v16h bf = frag_b(&Wl[nt * 16][ks * 32], 66, lane);
      acc = wmma32(a, bf, acc);
    }
    int o = nt * 16 + col;
    float bsv = bo[o];
#pragma unroll
    for (int v = 0; v < 8; ++v) {
      int r = r0 + mt * 16 + rr + v;
      x2t[r * 64 + o] = x1t[r * 64 + o] + acc[v] + bsv;
    }
  }
}

// ---------------- K8: LN2 (wave per row) -------------------------------------
__global__ void k_ln2(const float* __restrict__ x2t, const float* __restrict__ g,
                      const float* __restrict__ bta, float* __restrict__ xct) {
  const int w = threadIdx.x >> 5, lane = threadIdx.x & 31;
  const int r = blockIdx.x * 8 + w;
  float a  = x2t[r * 64 + lane];
  float b2 = x2t[r * 64 + 32 + lane];
  float m = wave_sum(a + b2) * (1.f / 64.f);
  float da = a - m, db = b2 - m;
  float var = wave_sum(da * da + db * db) * (1.f / 64.f);
  float rstd = rsqrtf(var + 1e-5f);
  xct[r * 64 + lane]      = da * rstd * g[lane]      + bta[lane];
  xct[r * 64 + 32 + lane] = db * rstd * g[32 + lane] + bta[32 + lane];
}

// ---------------- K9a: FFN layer 1 + ReLU (WMMA), f16 hidden -----------------
__global__ void k_ffn1(const float* __restrict__ xct, const float* __restrict__ w1,
                       const float* __restrict__ b1, _Float16* __restrict__ hbuf) {
  __shared__ _Float16 Al[64][66];
  __shared__ _Float16 Wl[256][66];
  const int r0 = blockIdx.x * 64;
  for (int idx = threadIdx.x; idx < 4096; idx += 256) {
    int o = idx >> 6, c = idx & 63;
    Al[o][c] = (_Float16)xct[(r0 + o) * 64 + c];
  }
  for (int idx = threadIdx.x; idx < 256 * 64; idx += 256) {
    int o = idx >> 6, c = idx & 63;
    Wl[o][c] = (_Float16)w1[c * 256 + o];
  }
  __syncthreads();
  const int w = threadIdx.x >> 5, lane = threadIdx.x & 31;
  const int rr = (lane >> 4) << 3, col = lane & 15;
  for (int tile = w; tile < 64; tile += 8) {
    int mt = tile >> 4, nt = tile & 15;
    v8f acc = {};
#pragma unroll
    for (int ks = 0; ks < 2; ++ks) {
      v16h a = frag_a(&Al[mt * 16][ks * 32], 66, lane);
      v16h bf = frag_b(&Wl[nt * 16][ks * 32], 66, lane);
      acc = wmma32(a, bf, acc);
    }
    int o = nt * 16 + col;
    float bsv = b1[o];
#pragma unroll
    for (int v = 0; v < 8; ++v)
      hbuf[(r0 + mt * 16 + rr + v) * 256 + o] = (_Float16)fmaxf(acc[v] + bsv, 0.f);
  }
}

// ------ K9b: FFN2 + residual + final transpose (double-buffered async) -------
__global__ void k_ffn2_out(const _Float16* __restrict__ hbuf, const float* __restrict__ xct,
                           const float* __restrict__ w2, const float* __restrict__ b2,
                           float* __restrict__ xout) {
  __shared__ _Float16 Al[2][64][32];
  __shared__ _Float16 Wl[64][258];
  const int r0 = blockIdx.x * 64;
  for (int idx = threadIdx.x; idx < 64 * 256; idx += 256) {
    int o = idx >> 8, c = idx & 255;
    Wl[o][c] = (_Float16)w2[c * 64 + o];
  }
  const int w = threadIdx.x >> 5, lane = threadIdx.x & 31;
  const int sr = threadIdx.x >> 2;
  const int sp = (threadIdx.x & 3) << 3;
  const unsigned ldsA[2] = { lds_addr_of(&Al[0][sr][sp]), lds_addr_of(&Al[1][sr][sp]) };
  const _Float16* gA = hbuf + ((r0 + sr) * 256 + sp);
  v8f acc[2] = {};
  async_b128(gA, ldsA[0]);
  for (int ks = 0; ks < 8; ++ks) {
    const int bf = ks & 1;
    if (ks < 7) {
      async_b128(gA + (ks + 1) * 32, ldsA[bf ^ 1]);
      wait_async1();         // oldest transfer (tile ks) complete
    } else {
      wait_async0();
    }
    __syncthreads();
#pragma unroll
    for (int tt = 0; tt < 2; ++tt) {
      int tile = w + tt * 8;
      int mt = tile >> 2, nt = tile & 3;
      v16h a = frag_a(&Al[bf][mt * 16][0], 32, lane);
      v16h bfrag = frag_b(&Wl[nt * 16][ks * 32], 258, lane);
      acc[tt] = wmma32(a, bfrag, acc[tt]);
    }
    __syncthreads();
  }
  const int rr = (lane >> 4) << 3, col = lane & 15;
#pragma unroll
  for (int tt = 0; tt < 2; ++tt) {
    int tile = w + tt * 8;
    int mt = tile >> 2, nt = tile & 3;
    int o = nt * 16 + col;
    float bsv = b2[o];
#pragma unroll
    for (int v = 0; v < 8; ++v) {
      int r = r0 + mt * 16 + rr + v;
      float val = xct[r * 64 + o] + acc[tt][v] + bsv;
      int b = r / 12288;
      int n = (r / 24) & 511;
      int t = r % 24;
      xout[((b * 64 + o) * 512 + n) * 24 + t] = val;
    }
  }
}

extern "C" void kernel_launch(void* const* d_in, const int* in_sizes, int n_in,
                              void* d_out, int out_size, void* d_ws, size_t ws_size,
                              hipStream_t stream) {
  (void)in_sizes; (void)n_in; (void)out_size; (void)ws_size;
  const float* x    = (const float*)d_in[0];
  const float* adj  = (const float*)d_in[1];
  const float* wq_s = (const float*)d_in[2];
  const float* bq_s = (const float*)d_in[3];
  const float* wk_s = (const float*)d_in[4];
  const float* bk_s = (const float*)d_in[5];
  const float* wv_s = (const float*)d_in[6];
  const float* bv_s = (const float*)d_in[7];
  const float* wq   = (const float*)d_in[8];
  const float* bq   = (const float*)d_in[9];
  const float* wk   = (const float*)d_in[10];
  const float* bk   = (const float*)d_in[11];
  const float* wv   = (const float*)d_in[12];
  const float* bv   = (const float*)d_in[13];
  const float* wo   = (const float*)d_in[14];
  const float* bo   = (const float*)d_in[15];
  const float* ln1g = (const float*)d_in[16];
  const float* ln1b = (const float*)d_in[17];
  const float* ln2g = (const float*)d_in[18];
  const float* ln2b = (const float*)d_in[19];
  const float* w1   = (const float*)d_in[20];
  const float* b1   = (const float*)d_in[21];
  const float* w2   = (const float*)d_in[22];
  const float* b2   = (const float*)d_in[23];

  float* xout = (float*)d_out;            // [B,C,N,T] 6291456 floats
  float* attn = xout + 6291456;           // [B,T,N,N] 50331648 floats

  // workspace layout (floats):
  float* ws  = (float*)d_ws;
  float* qc  = ws;                              //   786432  [B,8,T,N] f32
  float* kc  = qc + 786432;                     //   786432
  _Float16* vch = (_Float16*)(kc + 786432);     //  6291456 halves [B,64,T,N]
  float* xs  = (float*)(vch + 6291456);         //  6291456  [BT,64,N] (reuse: att)
  float* x1t = xs + 6291456;                    //  6291456  [R,64]   (reuse: xct)
  float* Z   = x1t + 6291456;                   // 25165824 floats big region
  _Float16* attn16 = (_Float16*)Z;              //  f16 attention copy (K2->K3)
  float* Qt  = Z;                               //  after K3, reuse as Qt/Kt/Vt
  float* Kt  = Qt + 6291456;
  float* Vt  = Kt + 6291456;
  float* att = xs;                              // temporal attn output
  float* x2t = Qt;                              // after K6, Qt region free
  float* xct = x1t;                             // after K7, x1t region free
  _Float16* hbuf = (_Float16*)Kt;               // 25165824 halves over Kt+Vt

  k_spatial_qkv<<<dim3(192, 4), 256, 0, stream>>>(x, wq_s, bq_s, wk_s, bk_s,
                                                  wv_s, bv_s, qc, kc, vch);
  k_spatial_attn<<<dim3(192, 64), 256, 0, stream>>>(qc, kc, adj, attn, attn16);
  k_spatial_av<<<dim3(8, 192), 256, 0, stream>>>(vch, attn16, xs);
  k_res_ln1<<<4096, 256, 0, stream>>>(x, xs, ln1g, ln1b, x1t);
  k_tqkv<<<1536, 256, 0, stream>>>(x1t, wq, bq, wk, bk, wv, bv, Qt, Kt, Vt);
  k_tattn<<<4096, 128, 0, stream>>>(Qt, Kt, Vt, att);
  k_wo_res<<<1536, 256, 0, stream>>>(att, x1t, wo, bo, x2t);
  k_ln2<<<12288, 256, 0, stream>>>(x2t, ln2g, ln2b, xct);
  k_ffn1<<<1536, 256, 0, stream>>>(xct, w1, b1, hbuf);
  k_ffn2_out<<<1536, 256, 0, stream>>>(hbuf, xct, w2, b2, xout);
}
